// GATLayer_27951647162472
// MI455X (gfx1250) — matrix operand order repaired and checked by reading, not compiled, not run on hardware
//
#include <hip/hip_runtime.h>

// ---------------------------------------------------------------------------
// GAT layer on MI455X (gfx1250), restructured as one HBM-bound WMMA GEMM:
//   C[8192,144] = adj[8192,8192](f16, exact for {0,1}) x B[8192,144](f16)
// where B[:,0:128] = exp(s_dst)*ft, B[:,128:130] = exp(s_dst), B[:,130:144]=0.
// Softmax normalization, self-loop and ELU are applied in the epilogue.
//
// Roofline: adj = 256 MB streamed once -> ~11 us at 23.3 TB/s; GEMM is only
// ~17 GFLOP in f16 WMMA, so the kernel is pure HBM-BW bound. adj loads are
// non-temporal so the reused B / Cacc working set stays resident in L2.
// ---------------------------------------------------------------------------

#define NN      8192
#define INDIM   64
#define HEADS   2
#define BCOLS   144          // 128 feature cols + 2 denom cols + 14 pad
#define NTILES  (BCOLS / 16) // 9 WMMA column tiles
#define KTILES  (NN / 32)    // 256 K-steps of 32
#define KSPLIT  8            // K-dimension split (atomic f32 accumulation)
#define MBLOCKS (NN / 128)   // 64 row blocks of 128 (8 waves x 16 rows)

typedef __attribute__((ext_vector_type(16))) _Float16 v16h;
typedef __attribute__((ext_vector_type(8)))  float    v8f;
typedef __attribute__((ext_vector_type(4)))  float    v4f;

// --- B-fragment swizzle (32x16 f16 B-matrix register layout, wave32) -------
// lanes 0-15 hold K=0..15 (n = lane), lanes 16-31 hold K=16..31 (n = lane-16),
// 16 halfs per lane, h = k & 15.  Stored so the GEMM reads one contiguous
// v16h (2 x b128) per lane per fragment.
__device__ __forceinline__ size_t swz_idx(int m, int c) {
  int kt   = m >> 5;
  int ko   = m & 31;
  int t    = c >> 4;
  int lane = (c & 15) + (((ko >> 4) & 1) << 4);
  int h    = ko & 15;
  return (((size_t)kt * NTILES + t) * 32 + lane) * 16 + h;
}

// --- zero-init of the f32 accumulator --------------------------------------
__global__ void gat_zero(float* __restrict__ p, int n) {
  int i = blockIdx.x * blockDim.x + threadIdx.x;
  if (i < n) p[i] = 0.0f;
}

// --- prep: per-node scalars + swizzled f16 B matrix ------------------------
// one block (128 threads) per node m.
__global__ void gat_prep(const float* __restrict__ feat,
                         const float* __restrict__ W_attn,
                         const float* __restrict__ b_attn,
                         const float* __restrict__ W_lin,
                         const float* __restrict__ b_lin,
                         _Float16* __restrict__ Bsw,
                         float* __restrict__ Cexp) {
  const int m   = blockIdx.x;
  const int tid = threadIdx.x;

  __shared__ float f[INDIM];
  __shared__ float g[HEADS];

  if (tid < INDIM) f[tid] = feat[(size_t)m * INDIM + tid];
  __syncthreads();

  if (tid < 4) {                       // tid 0,1: s_src; tid 2,3: s_dst
    const int  h   = tid & 1;
    const bool src = (tid < 2);
    const float* w = W_attn + h * (2 * INDIM) + (src ? 0 : INDIM);
    float s = 0.0f;
    #pragma unroll
    for (int k = 0; k < INDIM; ++k) s += f[k] * w[k];
    if (src) Cexp[m * HEADS + h] = expf(s + b_attn[h]);
    else     g[h]                = expf(s);
  }
  __syncthreads();

  // feature transform column c = tid (0..127)
  {
    const int c    = tid;
    const float* w = W_lin + (size_t)c * INDIM;
    float s = b_lin[c];
    #pragma unroll
    for (int k = 0; k < INDIM; ++k) s += f[k] * w[k];
    Bsw[swz_idx(m, c)] = (_Float16)(g[c >> 6] * s);
  }
  // denom columns 128,129 = g[h]; pad columns 130..143 = 0
  if (tid < 2)              Bsw[swz_idx(m, 128 + tid)] = (_Float16)g[tid];
  else if (tid < 16)        Bsw[swz_idx(m, 128 + tid)] = (_Float16)0.0f;
}

// --- main GEMM: C += adj x B via v_wmma_f32_16x16x32_f16 -------------------
// 256 threads = 8 waves; wave owns 16 rows x 144 cols (9 C fragments).
// A-fragment layout (16x32 f16): lane L holds row M=L&15; lanes<16 carry
// K {0..15}, lanes>=16 carry {16..31} -> one contiguous
// 64B run of the row: load fp32 straight from global (non-temporal),
// convert in VALU, no LDS and no barriers in the hot loop.
__global__ void __launch_bounds__(256)
gat_gemm(const float* __restrict__ adj,
         const _Float16* __restrict__ Bsw,
         float* __restrict__ Cacc) {
  const int lane = threadIdx.x & 31;
  const int wave = threadIdx.x >> 5;
  const int mblk = blockIdx.x % MBLOCKS;
  const int kspl = blockIdx.x / MBLOCKS;

  const int row0 = mblk * 128 + wave * 16;
  const int kt0  = kspl * (KTILES / KSPLIT);
  const int ktN  = kt0 + (KTILES / KSPLIT);

  const int arow = row0 + (lane & 15);
  const int koct = (lane >> 4) << 4;              // 0 or 16
  const float* aptr = adj + (size_t)arow * NN;

  v8f acc[NTILES];
  #pragma unroll
  for (int t = 0; t < NTILES; ++t) acc[t] = (v8f){0,0,0,0,0,0,0,0};

  for (int kt = kt0; kt < ktN; ++kt) {
    const int kb = kt * 32;

    // B fragments first: 9 x v16h into distinct registers so all 18 b128
    // loads clause together and the 9 WMMAs can issue back-to-back.
    const _Float16* bb = Bsw + ((size_t)kt * NTILES) * 512 + (size_t)lane * 16;
    v16h bfr[NTILES];
    #pragma unroll
    for (int t = 0; t < NTILES; ++t)
      bfr[t] = *(const v16h*)(bb + (size_t)t * 512);

    // A: 16 fp32 of this lane's row, one contiguous 64B run, streamed
    // non-temporally (adj is read exactly once -> keep L2 for B / Cacc).
    const v4f a0 = __builtin_nontemporal_load((const v4f*)(aptr + kb + koct));
    const v4f a1 = __builtin_nontemporal_load((const v4f*)(aptr + kb + koct + 4));
    const v4f a2 = __builtin_nontemporal_load((const v4f*)(aptr + kb + koct + 8));
    const v4f a3 = __builtin_nontemporal_load((const v4f*)(aptr + kb + koct + 12));

    v16h afrag;
    #pragma unroll
    for (int i = 0; i < 4; ++i) {
      afrag[i]      = (_Float16)a0[i];
      afrag[4 + i]  = (_Float16)a1[i];
      afrag[8 + i]  = (_Float16)a2[i];
      afrag[12 + i] = (_Float16)a3[i];
    }

    // 9 back-to-back WMMAs sharing one A fragment; reuse_a operand-cache
    // hint on all but the first (requires literal constants -> macro unroll).
#define GAT_WMMA(T, RA)                                                    \
    acc[T] = __builtin_amdgcn_wmma_f32_16x16x32_f16(                       \
        /*neg_a=*/false, afrag, /*neg_b=*/false, bfr[T],                   \
        /*c_mod=*/(short)0, acc[T], /*reuse_a=*/RA, /*reuse_b=*/false)
    GAT_WMMA(0, false);
    GAT_WMMA(1, true);
    GAT_WMMA(2, true);
    GAT_WMMA(3, true);
    GAT_WMMA(4, true);
    GAT_WMMA(5, true);
    GAT_WMMA(6, true);
    GAT_WMMA(7, true);
    GAT_WMMA(8, true);
#undef GAT_WMMA
  }

  // C fragment layout: lanes 0-15 N=lane, M=r; lanes 16-31 N=lane-16, M=r+8
  const int crow0 = row0 + ((lane >> 4) << 3);
  const int ccol  = lane & 15;
  #pragma unroll
  for (int t = 0; t < NTILES; ++t) {
    #pragma unroll
    for (int r = 0; r < 8; ++r) {
      atomicAdd(&Cacc[(size_t)(crow0 + r) * BCOLS + t * 16 + ccol], acc[t][r]);
    }
  }
}

// --- epilogue: self-loop, softmax normalize, ELU ---------------------------
__global__ void gat_epi(const float* __restrict__ Cacc,
                        const _Float16* __restrict__ Bsw,
                        const float* __restrict__ Cexp,
                        float* __restrict__ out) {
  const int idx = blockIdx.x * blockDim.x + threadIdx.x;   // 0 .. N*128-1
  const int n = idx >> 7;
  const int c = idx & 127;
  const int h = c >> 6;
  // adj_sl = adj + I  ->  C_full[n,*] = C_gemm[n,*] + 1.0 * B[n,*]
  const float num = Cacc[(size_t)n * BCOLS + c]
                  + (float)Bsw[swz_idx(n, c)];
  const float den = Cacc[(size_t)n * BCOLS + 128 + h]
                  + (float)Bsw[swz_idx(n, 128 + h)];
  const float ce = Cexp[n * HEADS + h];
  const float v  = (ce * num) / (ce * den + 1e-10f);
  out[idx] = v > 0.0f ? v : (expf(v) - 1.0f);
}

// ---------------------------------------------------------------------------
extern "C" void kernel_launch(void* const* d_in, const int* in_sizes, int n_in,
                              void* d_out, int out_size, void* d_ws, size_t ws_size,
                              hipStream_t stream) {
  (void)in_sizes; (void)n_in; (void)out_size; (void)ws_size;
  const float* adj    = (const float*)d_in[0];
  const float* feat   = (const float*)d_in[1];
  const float* W_attn = (const float*)d_in[2];
  const float* b_attn = (const float*)d_in[3];
  const float* W_lin  = (const float*)d_in[4];
  const float* b_lin  = (const float*)d_in[5];
  float* out = (float*)d_out;

  // workspace layout
  char* ws = (char*)d_ws;
  _Float16* Bsw  = (_Float16*)ws;                        // 8192*144 f16 = 2,359,296 B
  float*    Cexp = (float*)(ws + (size_t)NN * BCOLS * 2);            // 64 KB
  float*    Cacc = (float*)(ws + (size_t)NN * BCOLS * 2 + NN * HEADS * 4); // 4.5 MB

  const int cElems = NN * BCOLS;
  gat_zero<<<(cElems + 255) / 256, 256, 0, stream>>>(Cacc, cElems);
  gat_prep<<<NN, 128, 0, stream>>>(feat, W_attn, b_attn, W_lin, b_lin, Bsw, Cexp);
  gat_gemm<<<MBLOCKS * KSPLIT, 256, 0, stream>>>(adj, Bsw, Cacc);
  gat_epi<<<(NN * 128) / 256, 256, 0, stream>>>(Cacc, Bsw, Cexp, out);
}
